// BSRoformerAttention_54941221650513
// MI455X (gfx1250) — matrix operand order repaired and checked by reading, not compiled, not run on hardware
//
#include <hip/hip_runtime.h>
#include <math.h>

typedef unsigned short u16;
typedef unsigned int   u32;
typedef __attribute__((ext_vector_type(16))) __bf16 v16bf;
typedef __attribute__((ext_vector_type(8)))  float  v8f;

union Frag { v16bf v; uint4 u[2]; };
union Acc8 { v8f  v; float f[8]; };

#define S_LEN 2048
#define HIDN  1024
#define NH    16
#define NKV   4
#define HD    64
#define MROWS 8192          // B * S
#define ATT_SCALE 0.125f    // 1/sqrt(64)

__device__ __forceinline__ u16 f2bf(float f) {
  union { float f; u32 u; } x; x.f = f;
  u32 r = x.u + 0x7FFFu + ((x.u >> 16) & 1u);   // round-to-nearest-even
  return (u16)(r >> 16);
}

// CDNA5 async copy: memory -> LDS via the async engine (ASYNCcnt), 16B per lane.
// GV addressing: global_load_async_to_lds_b128 vdst_lds, v[addr:addr+1], off
__device__ __forceinline__ void async_b128(unsigned lds_addr, const void* g) {
  unsigned long long ga = (unsigned long long)g;
  asm volatile("global_load_async_to_lds_b128 %0, %1, off"
               :: "v"(lds_addr), "v"(ga) : "memory");
}
__device__ __forceinline__ void wait_async0() {
  asm volatile("s_wait_asynccnt 0x0" ::: "memory");
}
// low 32 bits of a flat shared pointer == wave-relative LDS byte address
__device__ __forceinline__ unsigned lds_addr_of(const void* p) {
  return (unsigned)(unsigned long long)p;
}

// ---------------------------------------------------------------- cast fp32 -> bf16
__global__ __launch_bounds__(256)
void cast_f32_bf16(const float* __restrict__ src, u16* __restrict__ dst, int n) {
  int i = blockIdx.x * 256 + threadIdx.x;
  if (i < n) dst[i] = f2bf(src[i]);
}

// ---------------------------------------------------------------- QKV GEMM + bias (+RoPE)
// C[m, h*64+d] = sum_k X[m,k] * W[h*64+d, k]  ; per-WG tile 64(M) x 64(N) = one head slice
// out layout: [B, nheads, S, 64]  (TRANS_OUT: [B, nheads, 64, S] for V)
template<bool ROPE, bool TRANS_OUT>
__global__ __launch_bounds__(128)
void qkv_gemm(const u16* __restrict__ X, const u16* __restrict__ W,
              const float* __restrict__ bias,
              const float* __restrict__ cosp, const float* __restrict__ sinp,
              u16* __restrict__ out, int nheads)
{
  __shared__ u16 As[2][64][32];   // double-buffered A tile (64 x 32 bf16)
  __shared__ u16 Ws[2][64][32];   // double-buffered W tile
  const int tid = threadIdx.x;
  const int w  = tid >> 5;          // wave id (wave32!)
  const int l  = tid & 31;
  const int hi = l >> 4, lo = l & 15;
  const int h = blockIdx.y;
  const int mbase = blockIdx.x * 64;

  Acc8 acc[4];
#pragma unroll
  for (int j = 0; j < 4; ++j)
#pragma unroll
    for (int i = 0; i < 8; ++i) acc[j].f[i] = 0.f;

  const int arow  = tid >> 1;
  const int ahalf = (tid & 1) * 16;
  const u16* Xrow = X + (size_t)(mbase + arow) * HIDN + ahalf;
  const u16* Wrow = W + ((size_t)h * 64 + arow) * HIDN + ahalf;

  // async-fill one k-tile (4 KB A + 4 KB W; 4x b128 per thread)
  auto issue_tile = [&](int kb, int buf) {
    async_b128(lds_addr_of(&As[buf][arow][ahalf]),     Xrow + kb);
    async_b128(lds_addr_of(&As[buf][arow][ahalf + 8]), Xrow + kb + 8);
    async_b128(lds_addr_of(&Ws[buf][arow][ahalf]),     Wrow + kb);
    async_b128(lds_addr_of(&Ws[buf][arow][ahalf + 8]), Wrow + kb + 8);
  };

  issue_tile(0, 0);
  int t = 0;
  for (int kb = 0; kb < HIDN; kb += 32, ++t) {
    wait_async0();        // own stage-t async loads landed in LDS
    __syncthreads();      // everyone's stage-t visible; everyone done with stage t-1
    if (kb + 32 < HIDN) issue_tile(kb + 32, (t + 1) & 1);  // overlap fill with WMMAs
    const int buf = t & 1;

    Frag a;
    {
      const int r  = w * 16 + lo;
      const int k8 = hi * 8;
      a.u[0] = *(const uint4*)&As[buf][r][k8];
      a.u[1] = *(const uint4*)&As[buf][r][k8 + 16];
    }
#pragma unroll
    for (int j = 0; j < 4; ++j) {
      Frag b;
      const int n   = j * 16 + lo;
      const int k16 = hi * 8;
      b.u[0] = *(const uint4*)&Ws[buf][n][k16];
      b.u[1] = *(const uint4*)&Ws[buf][n][k16 + 16];
      acc[j].v = __builtin_amdgcn_wmma_f32_16x16x32_bf16(
          false, a.v, false, b.v, (short)0, acc[j].v, false, false);
    }
  }

  float vals[4][8];
#pragma unroll
  for (int j = 0; j < 4; ++j) {
    const float bj = bias[h * 64 + j * 16 + lo];
#pragma unroll
    for (int i = 0; i < 8; ++i) vals[j][i] = acc[j].f[i] + bj;
  }

  if (ROPE) {
    // duplicated-half convention: cos/sin[d] == cos/sin[d+32]; pair tile j <-> j+2
#pragma unroll
    for (int i = 0; i < 8; ++i) {
      const int m = mbase + w * 16 + i + hi * 8;
      const int s = m & (S_LEN - 1);
#pragma unroll
      for (int j = 0; j < 2; ++j) {
        const int d0 = j * 16 + lo;              // 0..31
        const float c  = cosp[s * HD + d0];
        const float sn = sinp[s * HD + d0];
        const float x1 = vals[j][i];
        const float x2 = vals[j + 2][i];
        vals[j][i]     = x1 * c - x2 * sn;
        vals[j + 2][i] = x2 * c + x1 * sn;
      }
    }
  }

#pragma unroll
  for (int i = 0; i < 8; ++i) {
    const int m  = mbase + w * 16 + i + hi * 8;
    const int bb = m >> 11;            // /S
    const int s  = m & (S_LEN - 1);
#pragma unroll
    for (int j = 0; j < 4; ++j) {
      const int d = j * 16 + lo;
      const u16 ob = f2bf(vals[j][i]);
      if (TRANS_OUT)
        out[(((size_t)bb * nheads + h) * HD + d) * S_LEN + s] = ob;   // V^T
      else
        out[(((size_t)bb * nheads + h) * S_LEN + s) * HD + d] = ob;
    }
  }
}

// ---------------------------------------------------------------- flash attention
// Q [B,NH,S,64], K [B,NKV,S,64], Vt [B,NKV,64,S] -> O [B,S,NH*64] (bf16)
__global__ __launch_bounds__(128)
void attn_kernel(const u16* __restrict__ Q, const u16* __restrict__ K,
                 const u16* __restrict__ Vt, u16* __restrict__ O)
{
  __shared__ u16 Ks[2][64][64];   // [key][d]  : B-frag for Q*K^T contiguous in d
  __shared__ u16 Vs[2][64][64];   // [d][key]  : B-frag for P*V contiguous in key
  __shared__ u16 Ps[4][16][64];   // per-wave P tile [q][key]
  const int tid = threadIdx.x;
  const int w = tid >> 5, l = tid & 31, hi = l >> 4, lo = l & 15;
  const int bh = blockIdx.y;
  const int b = bh >> 4, h = bh & 15, kvh = h >> 2;   // GQA: 4 q-heads per kv-head
  const int qbase = blockIdx.x * 64;

  const u16* Qp = Q  + (((size_t)b * NH  + h)   * S_LEN) * HD;
  const u16* Kp = K  + (((size_t)b * NKV + kvh) * S_LEN) * HD;
  const u16* Vp = Vt + (((size_t)b * NKV + kvh) * HD) * S_LEN;

  // Q A-frags resident for the whole kernel (contraction over d: two K=32 chunks)
  Frag qa[2];
  {
    const int qrow = qbase + w * 16 + lo;
#pragma unroll
    for (int f = 0; f < 2; ++f) {
      const int k8 = f * 32 + hi * 8;
      qa[f].u[0] = *(const uint4*)(Qp + (size_t)qrow * HD + k8);
      qa[f].u[1] = *(const uint4*)(Qp + (size_t)qrow * HD + k8 + 16);
    }
  }

  float mrun[8], lrun[8];
  Acc8 oacc[4];
#pragma unroll
  for (int i = 0; i < 8; ++i) { mrun[i] = -3.0e30f; lrun[i] = 0.f; }
#pragma unroll
  for (int j = 0; j < 4; ++j)
#pragma unroll
    for (int i = 0; i < 8; ++i) oacc[j].f[i] = 0.f;

  const int crow  = tid >> 1;
  const int chalf = (tid & 1) * 32;

  // async-fill one 64-key chunk of K and V (8 KB each; 8x b128 per thread)
  auto issue_chunk = [&](int kc, int buf) {
    const u16* kg = Kp + (size_t)(kc + crow) * HD + chalf;
    const u16* vg = Vp + (size_t)crow * S_LEN + kc + chalf;
#pragma unroll
    for (int q4 = 0; q4 < 4; ++q4) {
      async_b128(lds_addr_of(&Ks[buf][crow][chalf + q4 * 8]), kg + q4 * 8);
      async_b128(lds_addr_of(&Vs[buf][crow][chalf + q4 * 8]), vg + q4 * 8);
    }
  };

  issue_chunk(0, 0);
  int t = 0;
  for (int kc = 0; kc < S_LEN; kc += 64, ++t) {
    wait_async0();        // own stage-t async loads landed
    __syncthreads();      // all waves: stage-t visible, stage t-1 compute finished
    if (kc + 64 < S_LEN) issue_chunk(kc + 64, (t + 1) & 1);  // overlap with compute
    const int buf = t & 1;

    // scores tile: 16 q rows x 64 keys, contraction over d=64 (2 WMMAs per subtile)
    Acc8 sc[4];
#pragma unroll
    for (int j = 0; j < 4; ++j) {
#pragma unroll
      for (int i = 0; i < 8; ++i) sc[j].f[i] = 0.f;
#pragma unroll
      for (int f = 0; f < 2; ++f) {
        Frag kb;
        const int key = j * 16 + lo;
        const int k16 = f * 32 + hi * 8;
        kb.u[0] = *(const uint4*)&Ks[buf][key][k16];
        kb.u[1] = *(const uint4*)&Ks[buf][key][k16 + 16];
        sc[j].v = __builtin_amdgcn_wmma_f32_16x16x32_bf16(
            false, qa[f].v, false, kb.v, (short)0, sc[j].v, false, false);
      }
    }
#pragma unroll
    for (int j = 0; j < 4; ++j)
#pragma unroll
      for (int i = 0; i < 8; ++i) sc[j].f[i] *= ATT_SCALE;

    // online softmax: row q = i + 8*hi, key cols spread over 16 lanes
#pragma unroll
    for (int i = 0; i < 8; ++i) {
      float t0 = fmaxf(fmaxf(sc[0].f[i], sc[1].f[i]), fmaxf(sc[2].f[i], sc[3].f[i]));
      t0 = fmaxf(t0, __shfl_xor(t0, 1, 32));
      t0 = fmaxf(t0, __shfl_xor(t0, 2, 32));
      t0 = fmaxf(t0, __shfl_xor(t0, 4, 32));
      t0 = fmaxf(t0, __shfl_xor(t0, 8, 32));
      const float mnew = fmaxf(mrun[i], t0);
      const float fac  = __expf(mrun[i] - mnew);
      mrun[i] = mnew;
      float p[4], rsum = 0.f;
#pragma unroll
      for (int j = 0; j < 4; ++j) { p[j] = __expf(sc[j].f[i] - mnew); rsum += p[j]; }
      rsum += __shfl_xor(rsum, 1, 32);
      rsum += __shfl_xor(rsum, 2, 32);
      rsum += __shfl_xor(rsum, 4, 32);
      rsum += __shfl_xor(rsum, 8, 32);
      lrun[i] = lrun[i] * fac + rsum;
#pragma unroll
      for (int j = 0; j < 4; ++j) {
        oacc[j].f[i] *= fac;
        Ps[w][i + hi * 8][j * 16 + lo] = f2bf(p[j]);  // C-layout -> [q][key] in LDS
      }
    }

    // out += P @ V  (contraction over 64 keys; wave-private Ps, in-order DS pipe)
#pragma unroll
    for (int f = 0; f < 2; ++f) {
      Frag pa;
      const int k8 = f * 32 + hi * 8;
      pa.u[0] = *(const uint4*)&Ps[w][lo][k8];
      pa.u[1] = *(const uint4*)&Ps[w][lo][k8 + 16];
#pragma unroll
      for (int j = 0; j < 4; ++j) {
        Frag vb;
        const int d   = j * 16 + lo;
        const int k16 = f * 32 + hi * 8;
        vb.u[0] = *(const uint4*)&Vs[buf][d][k16];
        vb.u[1] = *(const uint4*)&Vs[buf][d][k16 + 16];
        oacc[j].v = __builtin_amdgcn_wmma_f32_16x16x32_bf16(
            false, pa.v, false, vb.v, (short)0, oacc[j].v, false, false);
      }
    }
  }

  // normalize + store [B, S, NH*HD] bf16 for the output projection
#pragma unroll
  for (int i = 0; i < 8; ++i) {
    const float inv = 1.f / lrun[i];
    const int s = qbase + w * 16 + i + hi * 8;
#pragma unroll
    for (int j = 0; j < 4; ++j) {
      const int d = j * 16 + lo;
      O[((size_t)b * S_LEN + s) * (NH * HD) + h * HD + d] = f2bf(oacc[j].f[i] * inv);
    }
  }
}

// ---------------------------------------------------------------- output projection (fp32 out)
__global__ __launch_bounds__(128)
void proj_gemm(const u16* __restrict__ A, const u16* __restrict__ W,
               const float* __restrict__ bias, float* __restrict__ out)
{
  __shared__ u16 As[2][64][32];
  __shared__ u16 Ws[2][64][32];
  const int tid = threadIdx.x;
  const int w  = tid >> 5, l = tid & 31, hi = l >> 4, lo = l & 15;
  const int mbase = blockIdx.x * 64;
  const int nbase = blockIdx.y * 64;

  Acc8 acc[4];
#pragma unroll
  for (int j = 0; j < 4; ++j)
#pragma unroll
    for (int i = 0; i < 8; ++i) acc[j].f[i] = 0.f;

  const int arow  = tid >> 1;
  const int ahalf = (tid & 1) * 16;
  const u16* Arow = A + (size_t)(mbase + arow) * HIDN + ahalf;
  const u16* Wrow = W + (size_t)(nbase + arow) * HIDN + ahalf;

  auto issue_tile = [&](int kb, int buf) {
    async_b128(lds_addr_of(&As[buf][arow][ahalf]),     Arow + kb);
    async_b128(lds_addr_of(&As[buf][arow][ahalf + 8]), Arow + kb + 8);
    async_b128(lds_addr_of(&Ws[buf][arow][ahalf]),     Wrow + kb);
    async_b128(lds_addr_of(&Ws[buf][arow][ahalf + 8]), Wrow + kb + 8);
  };

  issue_tile(0, 0);
  int t = 0;
  for (int kb = 0; kb < HIDN; kb += 32, ++t) {
    wait_async0();
    __syncthreads();
    if (kb + 32 < HIDN) issue_tile(kb + 32, (t + 1) & 1);
    const int buf = t & 1;

    Frag a;
    {
      const int r  = w * 16 + lo;
      const int k8 = hi * 8;
      a.u[0] = *(const uint4*)&As[buf][r][k8];
      a.u[1] = *(const uint4*)&As[buf][r][k8 + 16];
    }
#pragma unroll
    for (int j = 0; j < 4; ++j) {
      Frag b;
      const int n   = j * 16 + lo;
      const int k16 = hi * 8;
      b.u[0] = *(const uint4*)&Ws[buf][n][k16];
      b.u[1] = *(const uint4*)&Ws[buf][n][k16 + 16];
      acc[j].v = __builtin_amdgcn_wmma_f32_16x16x32_bf16(
          false, a.v, false, b.v, (short)0, acc[j].v, false, false);
    }
  }

#pragma unroll
  for (int i = 0; i < 8; ++i) {
    const int m = mbase + w * 16 + i + hi * 8;
#pragma unroll
    for (int j = 0; j < 4; ++j) {
      const int n = nbase + j * 16 + lo;
      out[(size_t)m * HIDN + n] = acc[j].f[i] + bias[n];
    }
  }
}

// ---------------------------------------------------------------- launch
extern "C" void kernel_launch(void* const* d_in, const int* in_sizes, int n_in,
                              void* d_out, int out_size, void* d_ws, size_t ws_size,
                              hipStream_t stream) {
  const float* hs   = (const float*)d_in[0];
  const float* cosp = (const float*)d_in[1];
  const float* sinp = (const float*)d_in[2];
  const float* Wq   = (const float*)d_in[3];
  const float* bq   = (const float*)d_in[4];
  const float* Wk   = (const float*)d_in[5];
  const float* bk   = (const float*)d_in[6];
  const float* Wv   = (const float*)d_in[7];
  const float* bv   = (const float*)d_in[8];
  const float* Wo   = (const float*)d_in[9];
  const float* bo   = (const float*)d_in[10];

  char* ws = (char*)d_ws;                 // ~61 MB of bf16 staging
  u16* Xb  = (u16*)(ws);                  // [8192,1024]
  u16* Wqb = (u16*)(ws + 16777216);       // [1024,1024]
  u16* Wkb = (u16*)(ws + 18874368);       // [256,1024]
  u16* Wvb = (u16*)(ws + 19398656);       // [256,1024]
  u16* Wob = (u16*)(ws + 19922944);       // [1024,1024]
  u16* Qb  = (u16*)(ws + 22020096);       // [4,16,2048,64]
  u16* Kb  = (u16*)(ws + 38797312);       // [4,4,2048,64]
  u16* Vtb = (u16*)(ws + 42991616);       // [4,4,64,2048]  (pre-transposed)
  u16* Ab  = (u16*)(ws + 47185920);       // [8192,1024] attention output

  cast_f32_bf16<<<8388608/256, 256, 0, stream>>>(hs, Xb, 8388608);
  cast_f32_bf16<<<1048576/256, 256, 0, stream>>>(Wq, Wqb, 1048576);
  cast_f32_bf16<<< 262144/256, 256, 0, stream>>>(Wk, Wkb, 262144);
  cast_f32_bf16<<< 262144/256, 256, 0, stream>>>(Wv, Wvb, 262144);
  cast_f32_bf16<<<1048576/256, 256, 0, stream>>>(Wo, Wob, 1048576);

  qkv_gemm<true,  false><<<dim3(128, 16), 128, 0, stream>>>(Xb, Wqb, bq, cosp, sinp, Qb,  NH);
  qkv_gemm<true,  false><<<dim3(128,  4), 128, 0, stream>>>(Xb, Wkb, bk, cosp, sinp, Kb,  NKV);
  qkv_gemm<false, true ><<<dim3(128,  4), 128, 0, stream>>>(Xb, Wvb, bv, cosp, sinp, Vtb, NKV);

  attn_kernel<<<dim3(32, 64), 128, 0, stream>>>(Qb, Kb, Vtb, Ab);

  proj_gemm<<<dim3(128, 16), 128, 0, stream>>>(Ab, Wob, bo, (float*)d_out);
}
